// Model_81956565942963
// MI455X (gfx1250) — matrix operand order repaired and checked by reading, not compiled, not run on hardware
//
#include <hip/hip_runtime.h>

// ---------------- model constants ----------------
#define BB   16
#define LL   512
#define NN   64
#define DD   128
#define SS   32          // LL / GRAPH_SCALE
#define GG   16          // GRAPH_SCALE
#define RR   8           // GRAPH_RANK
#define PP   96          // PRED_LEN
#define BN   (BB*NN)     // 1024
#define ROWD ((size_t)LL*DD)   // 65536 elems per (b,n) series
#define KCHUNKS 16       // head GEMM K-split

// ---------------- WMMA types ----------------
typedef __attribute__((ext_vector_type(16))) __bf16 v16bf;
typedef __attribute__((ext_vector_type(8)))  float  v8f;
typedef unsigned int u32x4 __attribute__((ext_vector_type(4)));
typedef int          i32x4 __attribute__((ext_vector_type(4)));
typedef int          i32x8 __attribute__((ext_vector_type(8)));

union Frag16 {
    v16bf v;
    uint4 q[2];
    unsigned short s[16];
};

__device__ __forceinline__ unsigned short f2bf(float f) {
    unsigned u = __float_as_uint(f);
    u += 0x7FFFu + ((u >> 16) & 1u);      // round-to-nearest-even
    return (unsigned short)(u >> 16);
}

__device__ __forceinline__ v8f wmma_bf16(const Frag16& a, const Frag16& b, v8f c) {
    return __builtin_amdgcn_wmma_f32_16x16x32_bf16(
        /*neg_a*/false, a.v, /*neg_b*/false, b.v,
        /*c_mod*/(short)0, c, /*reuse_a*/false, /*reuse_b*/false);
}

__device__ __forceinline__ float gelu_tanh(float y) {
    const float c0 = 0.7978845608028654f, c1 = 0.044715f;
    return 0.5f * y * (1.0f + tanhf(c0 * (y + c1 * y * y * y)));
}

// ---------------- Tensor Data Mover: 2D bf16 tile -> LDS ----------------
// Builds D# per CDNA5 ISA and issues
// TENSOR_LOAD_TO_LDS (TENSORcnt). Caller: wave0 only, then wait+barrier.
#if __has_builtin(__builtin_amdgcn_tensor_load_to_lds)
#define HAVE_TDM 1
#else
#define HAVE_TDM 0
#endif

#if HAVE_TDM
__device__ __forceinline__ void tdm_load_bf16_2d(unsigned lds_byte_addr, const void* gsrc,
                                                 unsigned tile_w, unsigned tile_h,
                                                 unsigned stride_elems) {
    unsigned long long ga = (unsigned long long)gsrc;
    // group0: [1:0]=count=1 | lds_addr | global_addr[56:0] | type=2 in [127:126]
    u32x4 g0;
    g0[0] = 1u;
    g0[1] = lds_byte_addr;
    g0[2] = (unsigned)(ga & 0xFFFFFFFFu);
    g0[3] = (unsigned)((ga >> 32) & 0x01FFFFFFu) | 0x80000000u;
    // group1: data_size=1 (2B); tensor_dim0=tile_w; tensor_dim1=tile_h;
    //         tile_dim0=tile_w; tile_dim1=tile_h; dim0_stride=stride_elems
    i32x8 g1;
    g1[0] = 0x00010000;               // [17:16] data_size = 1 -> 2 bytes
    g1[1] = (int)(tile_w << 16);      // tensor_dim0[15:0] @ bits[79:64... word1 31:16]
    g1[2] = (int)(tile_h << 16);      // tensor_dim0 hi=0 ; tensor_dim1[15:0]
    g1[3] = (int)(tile_w << 16);      // tensor_dim1 hi=0 ; tile_dim0
    g1[4] = (int)tile_h;              // tile_dim1 ; tile_dim2=0
    g1[5] = (int)stride_elems;        // tensor_dim0_stride[31:0]
    g1[6] = 0;
    g1[7] = 0;
    i32x4 z4 = {0, 0, 0, 0};
#if defined(__clang_major__) && __clang_major__ >= 23
    i32x8 z8 = {0, 0, 0, 0, 0, 0, 0, 0};
    __builtin_amdgcn_tensor_load_to_lds(g0, g1, z4, z4, z8, 0);
#else
    __builtin_amdgcn_tensor_load_to_lds(g0, g1, z4, z4, 0);
#endif
}
#endif

// ---------------- pack f32 [K][N] -> bf16 WMMA B-fragment order ----------------
__global__ __launch_bounds__(256)
void k_pack_b(const float* __restrict__ src, unsigned short* __restrict__ dst,
              int K, int N, int total_threads) {
    (void)K;
    int gid = blockIdx.x * 256 + threadIdx.x;
    if (gid >= total_threads) return;
    int fid  = gid >> 5;
    int lane = gid & 31;
    int nt   = fid % (N >> 4);
    int kt   = fid / (N >> 4);
    int n    = lane & 15;
    int half = lane >> 4;
    #pragma unroll
    for (int j = 0; j < 16; ++j) {
        int k = kt * 32 + half * 16 + j;
        dst[(size_t)gid * 16 + j] = f2bf(src[(size_t)k * N + nt * 16 + n]);
    }
}

// ---------------- embedding: h = x * w_emb + b_emb (f32 + bf16 mirror) ----------
__global__ __launch_bounds__(256)
void k_embed(const float* __restrict__ x, const float* __restrict__ wemb,
             const float* __restrict__ bemb, float* __restrict__ h,
             unsigned short* __restrict__ hbf) {
    size_t gid = (size_t)blockIdx.x * 256 + threadIdx.x;   // BN*LL*DD threads
    int d  = (int)(gid & 127);
    int t  = (int)((gid >> 7) & 511);
    int bn = (int)(gid >> 16);
    int b = bn >> 6, n = bn & 63;
    float v = x[((size_t)b * LL + t) * NN + n] * wemb[d] + bemb[d];
    h[gid]   = v;
    hbf[gid] = f2bf(v);
}

// ---------------- dilated causal conv layer: gelu(conv(h)) + h -----------------
// one block = one (bn, 16-token) tile; 8 waves cover 128 output channels.
// A tile (20 rows x 128 ch bf16) staged via TDM (fallback: manual loads).
__global__ __launch_bounds__(256)
void k_tcn(const unsigned short* __restrict__ hbf_in, const float* __restrict__ h_in,
           const unsigned short* __restrict__ wpak, const float* __restrict__ bias,
           float* __restrict__ h_out, unsigned short* __restrict__ hbf_out, int dil) {
    __shared__ __align__(16) unsigned short hs[20 * DD];   // rows t0-4 .. t0+15
    int bn = blockIdx.x >> 5;
    int t0 = (blockIdx.x & 31) << 4;
    int tid = threadIdx.x;

#if HAVE_TDM
    if (t0 == 0) {                       // causal left pad: zero rows 0..3
        for (int i = tid; i < 4 * DD; i += 256) hs[i] = 0;
    }
    if (tid < 32) {                      // wave0 issues the TDM tile load
        int rstart = (t0 == 0) ? 0 : (t0 - 4);
        unsigned nrows = (t0 == 0) ? 16u : 20u;
        unsigned ldsoff = (unsigned)(size_t)hs + ((t0 == 0) ? 4u * DD * 2u : 0u);
        tdm_load_bf16_2d(ldsoff,
                         hbf_in + ((size_t)bn * LL + rstart) * DD,
                         DD, nrows, DD);
        __builtin_amdgcn_s_wait_tensorcnt(0);
    }
    __syncthreads();
#else
    for (int i = tid; i < 20 * DD; i += 256) {
        int rr = i >> 7, ch = i & 127;
        int t = t0 + rr - 4;
        hs[i] = (t >= 0) ? hbf_in[((size_t)bn * LL + t) * DD + ch] : (unsigned short)0;
    }
    __syncthreads();
#endif

    int wave = tid >> 5, lane = tid & 31;
    int half = lane >> 4, mrow = lane & 15;
    v8f acc = {};
    #pragma unroll
    for (int k = 0; k < 3; ++k) {
        int ro = 4 - k * dil;                       // LDS row offset for this tap
        const unsigned short* wm = wpak + (size_t)k * (32 * 512);
        #pragma unroll
        for (int kt = 0; kt < 4; ++kt) {
            Frag16 a, bw;
            int row = mrow + ro;
            int k0  = kt * 32 + half * 16;
            a.q[0] = *(const uint4*)&hs[row * DD + k0];
            a.q[1] = *(const uint4*)&hs[row * DD + k0 + 8];
            const uint4* bq = (const uint4*)wm + ((size_t)(kt * 8 + wave) * 32 + lane) * 2;
            bw.q[0] = bq[0];
            bw.q[1] = bq[1];
            acc = wmma_bf16(a, bw, acc);
        }
    }
    int co = wave * 16 + (lane & 15);
    #pragma unroll
    for (int v = 0; v < 8; ++v) {
        int m = v + half * 8;
        size_t idx = ((size_t)bn * LL + (t0 + m)) * DD + co;
        float y = acc[v] + bias[co];
        float o = gelu_tanh(y) + h_in[idx];
        h_out[idx]   = o;
        hbf_out[idx] = f2bf(o);
    }
}

// ---------------- graph generation: z -> e1,e2 -> scores -> softmax(adj) -------
__global__ __launch_bounds__(256)
void k_graph(const float* __restrict__ h,
             const float* __restrict__ w1, const float* __restrict__ b1,
             const float* __restrict__ w2, const float* __restrict__ b2,
             float* __restrict__ adjf, unsigned short* __restrict__ adjbf) {
    __shared__ float zs[NN * DD];          // 32 KB; reused for scores after sync
    __shared__ float e1s[NN * RR], e2s[NN * RR];
    int b = blockIdx.x >> 5, s = blockIdx.x & 31;
    int tid = threadIdx.x;

    for (int i = tid; i < NN * DD; i += 256) {            // segment means
        int n = i >> 7, d = i & 127;
        const float* p = h + (((size_t)(b * NN + n) * LL + s * GG) * DD + d);
        float sum = 0.f;
        #pragma unroll
        for (int g = 0; g < GG; ++g) sum += p[(size_t)g * DD];
        zs[i] = sum * (1.0f / GG);
    }
    __syncthreads();
    for (int i = tid; i < NN * RR; i += 256) {            // low-rank projections
        int n = i >> 3, r = i & 7;
        float a1 = b1[r], a2 = b2[r];
        for (int d = 0; d < DD; ++d) {
            float zv = zs[n * DD + d];
            a1 += zv * w1[d * RR + r];
            a2 += zv * w2[d * RR + r];
        }
        e1s[i] = a1; e2s[i] = a2;
    }
    __syncthreads();
    float* scs = zs;                                      // alias (z dead)
    const float inv = 0.35355339059327373f;               // 1/sqrt(8)
    for (int i = tid; i < NN * NN; i += 256) {
        int n = i >> 6, m = i & 63;
        float d0 = 0.f;
        #pragma unroll
        for (int r = 0; r < RR; ++r) d0 += e1s[n * RR + r] * e2s[m * RR + r];
        scs[i] = d0 * inv;
    }
    __syncthreads();
    if (tid < NN) {                                       // row softmax
        int n = tid;
        float mx = -1e30f;
        for (int m = 0; m < NN; ++m) mx = fmaxf(mx, scs[n * NN + m]);
        float sum = 0.f;
        for (int m = 0; m < NN; ++m) sum += expf(scs[n * NN + m] - mx);
        float rinv = 1.0f / sum;
        size_t base = ((size_t)(b * SS + s) * NN + n) * NN;
        for (int m = 0; m < NN; ++m) {
            float a = expf(scs[n * NN + m] - mx) * rinv;
            adjf[base + m]  = a;
            adjbf[base + m] = f2bf(a);
        }
    }
}

// ---------------- zero the reg scalar slot ----------------
__global__ void k_zero(float* p) { if (threadIdx.x == 0 && blockIdx.x == 0) *p = 0.f; }

// ---------------- adjacency smoothness regularizer ----------------
__global__ __launch_bounds__(256)
void k_reg(const float* __restrict__ adjf, float* __restrict__ reg) {
    __shared__ float red[256];
    int s = blockIdx.x + 1;                                 // s = 1..SS-1
    float p = 0.f;
    for (int i = threadIdx.x; i < BB * NN * NN; i += 256) {
        int b = i >> 12, rem = i & 4095;
        size_t o1 = (size_t)(b * SS + s) * 4096 + rem;
        size_t o0 = (size_t)(b * SS + s - 1) * 4096 + rem;
        p += fabsf(adjf[o1] - adjf[o0]);
    }
    red[threadIdx.x] = p;
    __syncthreads();
    for (int off = 128; off > 0; off >>= 1) {
        if (threadIdx.x < off) red[threadIdx.x] += red[threadIdx.x + off];
        __syncthreads();
    }
    if (threadIdx.x == 0) atomicAdd(reg, red[0] * (1.0f / (BB * NN * NN)));
}

// ---------------- graph mixing: h_mix = adj @ h_seg (per b,s) ------------------
__global__ __launch_bounds__(256)
void k_mix(const unsigned short* __restrict__ hbf, const unsigned short* __restrict__ adjbf,
           unsigned short* __restrict__ hmix) {
    __shared__ __align__(16) unsigned short adjs[NN * NN];
    int b = blockIdx.x >> 5, s = blockIdx.x & 31;
    int tid = threadIdx.x;
    const unsigned short* asrc = adjbf + (size_t)(b * SS + s) * (NN * NN);

#if HAVE_TDM
    if (tid < 32) {                      // wave0: TDM linear 8KB adjacency tile
        tdm_load_bf16_2d((unsigned)(size_t)adjs, asrc, NN * NN, 1, NN * NN);
        __builtin_amdgcn_s_wait_tensorcnt(0);
    }
    __syncthreads();
#else
    for (int i = tid; i < NN * NN; i += 256) adjs[i] = asrc[i];
    __syncthreads();
#endif

    int wave = tid >> 5, lane = tid & 31, half = lane >> 4;
    for (int ci = 0; ci < 16; ++ci) {
        int ct = ci * 8 + wave;                   // column tile of 2048 = G*D
        int c  = ct * 16 + (lane & 15);
        int g = c >> 7, d = c & 127;
        size_t coloff = (size_t)(s * GG + g) * DD + d;
        Frag16 b0, b1;
        #pragma unroll
        for (int j = 0; j < 16; ++j) {           // K rows = series index m
            int m0 = half * 16 + j;
            b0.s[j] = hbf[(size_t)(b * NN + m0)      * ROWD + coloff];
            b1.s[j] = hbf[(size_t)(b * NN + 32 + m0) * ROWD + coloff];
        }
        #pragma unroll
        for (int mt = 0; mt < 4; ++mt) {
            Frag16 a0, a1;
            int row = mt * 16 + (lane & 15);
            a0.q[0] = *(const uint4*)&adjs[row * NN + half * 16];
            a0.q[1] = *(const uint4*)&adjs[row * NN + half * 16 + 8];
            a1.q[0] = *(const uint4*)&adjs[row * NN + 32 + half * 16];
            a1.q[1] = *(const uint4*)&adjs[row * NN + 32 + half * 16 + 8];
            v8f acc = {};
            acc = wmma_bf16(a0, b0, acc);
            acc = wmma_bf16(a1, b1, acc);
            #pragma unroll
            for (int v = 0; v < 8; ++v) {
                int outn = mt * 16 + v + half * 8;
                hmix[(size_t)(b * NN + outn) * ROWD + coloff] = f2bf(acc[v]);
            }
        }
    }
}

// ---------------- head GEMM (K-split): part[c][row][p] = A_chunk @ W_chunk -----
// grid = 64 row-tiles x KCHUNKS; 6 waves cover 96 pred cols; 128 WMMAs each.
__global__ __launch_bounds__(192)
void k_head(const unsigned short* __restrict__ hmix, const unsigned short* __restrict__ wpak,
            float* __restrict__ part) {
    int wave = threadIdx.x >> 5, lane = threadIdx.x & 31, half = lane >> 4;
    int mt    = blockIdx.x & 63;
    int chunk = blockIdx.x >> 6;
    int r0 = mt * 16;
    const unsigned short* rowp = hmix + (size_t)(r0 + (lane & 15)) * ROWD;
    v8f acc = {};
    int ktEnd = (chunk + 1) * (2048 / KCHUNKS);
    for (int kt = chunk * (2048 / KCHUNKS); kt < ktEnd; ++kt) {
        Frag16 a, bw;
        int k0 = kt * 32 + half * 16;
        a.q[0] = *(const uint4*)&rowp[k0];
        a.q[1] = *(const uint4*)&rowp[k0 + 8];
        const uint4* bq = (const uint4*)wpak + ((size_t)(kt * 6 + wave) * 32 + lane) * 2;
        bw.q[0] = bq[0];
        bw.q[1] = bq[1];
        acc = wmma_bf16(a, bw, acc);
    }
    int p = wave * 16 + (lane & 15);
    #pragma unroll
    for (int v = 0; v < 8; ++v) {
        int row = r0 + v + half * 8;
        part[(size_t)chunk * (BN * PP) + (size_t)row * PP + p] = acc[v];
    }
}

// ---------------- head reduction + bias + transpose to [B, P, N] ---------------
__global__ __launch_bounds__(256)
void k_head_red(const float* __restrict__ part, const float* __restrict__ head_b,
                float* __restrict__ out) {
    int gid = blockIdx.x * 256 + threadIdx.x;          // BN*PP threads
    int p = gid % PP, row = gid / PP;
    float s = head_b[p];
    #pragma unroll
    for (int c = 0; c < KCHUNKS; ++c) s += part[(size_t)c * (BN * PP) + (size_t)row * PP + p];
    int b = row >> 6, n = row & 63;
    out[((size_t)b * PP + p) * NN + n] = s;
}

// =====================================================================
extern "C" void kernel_launch(void* const* d_in, const int* in_sizes, int n_in,
                              void* d_out, int out_size, void* d_ws, size_t ws_size,
                              hipStream_t stream) {
    (void)in_sizes; (void)n_in; (void)out_size; (void)ws_size;

    const float* x_enc  = (const float*)d_in[0];
    const float* w_emb  = (const float*)d_in[4];
    const float* b_emb  = (const float*)d_in[5];
    const float* conv_w = (const float*)d_in[6];
    const float* conv_b = (const float*)d_in[7];
    const float* gg_w1  = (const float*)d_in[8];
    const float* gg_b1  = (const float*)d_in[9];
    const float* gg_w2  = (const float*)d_in[10];
    const float* gg_b2  = (const float*)d_in[11];
    const float* head_w = (const float*)d_in[12];
    const float* head_b = (const float*)d_in[13];
    float* out = (float*)d_out;

    // workspace carve-up (256B-aligned)
    char* ws = (char*)d_ws;
    size_t off = 0;
    auto carve = [&](size_t bytes) { size_t o = off; off = (off + bytes + 255) & ~(size_t)255; return o; };
    const size_t ACT_F32 = (size_t)BN * ROWD * 4;          // 256 MB
    const size_t ACT_BF  = (size_t)BN * ROWD * 2;          // 128 MB
    float*          hA     = (float*)         (ws + carve(ACT_F32));
    float*          hB     = (float*)         (ws + carve(ACT_F32));
    unsigned short* hbfA   = (unsigned short*)(ws + carve(ACT_BF));
    unsigned short* hbfB   = (unsigned short*)(ws + carve(ACT_BF));
    unsigned short* hmix   = (unsigned short*)(ws + carve(ACT_BF));
    unsigned short* convp  = (unsigned short*)(ws + carve((size_t)6 * 32 * 512 * 2));
    unsigned short* headp  = (unsigned short*)(ws + carve((size_t)12288 * 512 * 2));
    float*          adjf   = (float*)         (ws + carve((size_t)BB * SS * NN * NN * 4));
    unsigned short* adjbf  = (unsigned short*)(ws + carve((size_t)BB * SS * NN * NN * 2));
    float*          hpart  = (float*)         (ws + carve((size_t)KCHUNKS * BN * PP * 4));

    // 1) pack weights into WMMA B-fragment order (bf16)
    for (int lk = 0; lk < 6; ++lk)       // conv: 2 layers x 3 taps of [128][128]
        k_pack_b<<<4, 256, 0, stream>>>(conv_w + (size_t)lk * DD * DD,
                                        convp + (size_t)lk * 32 * 512,
                                        DD, DD, 32 * 32);
    k_pack_b<<<1536, 256, 0, stream>>>(head_w, headp, (int)ROWD, PP, 12288 * 32);

    // 2) value embedding
    k_embed<<<262144, 256, 0, stream>>>(x_enc, w_emb, b_emb, hA, hbfA);

    // 3) TCN: layer0 (dil=1) hA->hB, layer1 (dil=2) hB->hA
    k_tcn<<<32768, 256, 0, stream>>>(hbfA, hA, convp,                conv_b,      hB, hbfB, 1);
    k_tcn<<<32768, 256, 0, stream>>>(hbfB, hB, convp + 3 * 32 * 512, conv_b + DD, hA, hbfA, 2);

    // 4) graph generation + softmax adjacency
    k_graph<<<BB * SS, 256, 0, stream>>>(hA, gg_w1, gg_b1, gg_w2, gg_b2, adjf, adjbf);

    // 5) smoothness regularizer -> d_out[B*P*N]
    k_zero<<<1, 1, 0, stream>>>(out + (size_t)BB * PP * NN);
    k_reg<<<SS - 1, 256, 0, stream>>>(adjf, out + (size_t)BB * PP * NN);

    // 6) graph-weighted mixing (bf16 WMMA)
    k_mix<<<BB * SS, 256, 0, stream>>>(hbfA, adjbf, hmix);

    // 7) head GEMM, K-split across 1024 blocks, then deterministic reduction
    k_head<<<64 * KCHUNKS, 192, 0, stream>>>(hmix, headp, hpart);
    k_head_red<<<(BN * PP) / 256, 256, 0, stream>>>(hpart, head_b, out);
}
